// BasicGNNEncoder_22325240004850
// MI455X (gfx1250) — matrix operand order repaired and checked by reading, not compiled, not run on hardware
//
#include <hip/hip_runtime.h>
#include <hip/hip_bf16.h>

// ---------------------------------------------------------------------------
// Types for CDNA5 WMMA (wave32, v_wmma_f32_16x16x32_bf16)
// ---------------------------------------------------------------------------
typedef __attribute__((ext_vector_type(16))) __bf16         v16bf;
typedef __attribute__((ext_vector_type(8)))  float          v8f;

#define WMMA_BF16(A, B, C) \
  __builtin_amdgcn_wmma_f32_16x16x32_bf16(false, (A), false, (B), (short)0, (C), false, false)

#define N_NODES 100000
#define N_EDGES 640000
#define FEAT    256
#define HID     128

typedef unsigned short u16;

// f32 -> bf16 (round-to-nearest-even), used only in epilogues / cvt passes
__device__ __forceinline__ u16 f2bf(float f) {
  union { float f; unsigned u; } c; c.f = f;
  unsigned u = c.u;
  u += 0x7fffu + ((u >> 16) & 1u);
  return (u16)(u >> 16);
}

// A fragment (16x32 bf16, MxK): lane supplies row (lane&15),
// 16 contiguous K at kb=(lane>=16)?16:0. p = row + k0 + kb.
// Pure memory: two 16-byte loads, no VALU.
__device__ __forceinline__ v16bf load_a_bf(const u16* __restrict__ p) {
  union { v16bf b; uint4 q[2]; } cv;
  cv.q[0] = *(const uint4*)(p);
  cv.q[1] = *(const uint4*)(p + 8);
  return cv.b;
}

// B fragment (32x16 bf16, KxN) for C = A*W^T: lane supplies col (lane&15),
// 16 contiguous K at kb=(lane>=16)?16:0. p = W_row(col) + k0 + kb.
__device__ __forceinline__ v16bf load_b_bf(const u16* __restrict__ p) {
  union { v16bf b; uint4 q[2]; } cv;
  cv.q[0] = *(const uint4*)(p);
  cv.q[1] = *(const uint4*)(p + 8);
  return cv.b;
}

// ---------------------------------------------------------------------------
// Generic f32 -> bf16 pass (n4 = count/4)
// ---------------------------------------------------------------------------
__global__ void cvt_kernel(const float* __restrict__ src, u16* __restrict__ dst, int n4) {
  int i = blockIdx.x * blockDim.x + threadIdx.x;
  if (i < n4) {
    float4 v = ((const float4*)src)[i];
    ushort4 o;
    o.x = f2bf(v.x); o.y = f2bf(v.y); o.z = f2bf(v.z); o.w = f2bf(v.w);
    ((ushort4*)dst)[i] = o;
  }
}

__global__ void zero_kernel(float* __restrict__ p, int n) {
  int i = blockIdx.x * blockDim.x + threadIdx.x;
  if (i < n) p[i] = 0.f;
}

// ---------------------------------------------------------------------------
// h = relu(Xb @ wpb^T + bp); writes f32 h and bf16 hb
// grid 6250 x 256 (8 waves; wave w -> cols 16w..16w+15; block -> 16 rows)
// ---------------------------------------------------------------------------
__global__ __launch_bounds__(256) void proj_kernel(
    const u16* __restrict__ Xb, const u16* __restrict__ Wpb,
    const float* __restrict__ bp, float* __restrict__ H, u16* __restrict__ Hb) {
  const int lane = threadIdx.x & 31;
  const int wv   = threadIdx.x >> 5;
  const int m0   = blockIdx.x << 4;
  const int n0   = wv << 4;
  const int hi   = lane >> 4;
  const int l15  = lane & 15;
  const size_t arow = (size_t)(m0 + l15) * FEAT + (hi << 4);
  const size_t brow = (size_t)(n0 + l15) * FEAT + (hi << 4);
  v8f acc = {};
#pragma unroll
  for (int k = 0; k < FEAT; k += 32) {
    v16bf a = load_a_bf(Xb + arow + k);
    v16bf b = load_b_bf(Wpb + brow + k);
    acc = WMMA_BF16(a, b, acc);
  }
  const int   col   = n0 + l15;
  const float bias  = bp[col];
  const int   rbase = m0 + (hi << 3);
#pragma unroll
  for (int r = 0; r < 8; ++r) {
    float v = acc[r] + bias;
    v = v > 0.f ? v : 0.f;
    const size_t o = (size_t)(rbase + r) * HID + col;
    H[o]  = v;
    Hb[o] = f2bf(v);
  }
}

// ---------------------------------------------------------------------------
// deg[n] = #edges with dst == n
// ---------------------------------------------------------------------------
__global__ void degree_kernel(const int* __restrict__ dst, float* __restrict__ deg, int n) {
  int e = blockIdx.x * blockDim.x + threadIdx.x;
  if (e < n) atomicAdd(deg + dst[e], 1.0f);
}

// ---------------------------------------------------------------------------
// agg[dst[e]] += h[src[e]] — one wave per edge, float4 per lane (32*4 = 128)
// f32 accumulation via L2 atomics (h/agg are L2-resident on 192MB L2)
// ---------------------------------------------------------------------------
__global__ __launch_bounds__(256) void scatter_kernel(
    const float* __restrict__ H, const int* __restrict__ src,
    const int* __restrict__ dst, float* __restrict__ agg, int nE) {
  const int wave = (int)((blockIdx.x * blockDim.x + threadIdx.x) >> 5);
  const int lane = threadIdx.x & 31;
  if (wave >= nE) return;
  const int s = src[wave];
  const int d = dst[wave];
  const float4 v = ((const float4*)(H + (size_t)s * HID))[lane];
  float* o = agg + (size_t)d * HID + lane * 4;
  atomicAdd(o + 0, v.x);
  atomicAdd(o + 1, v.y);
  atomicAdd(o + 2, v.z);
  atomicAdd(o + 3, v.w);
}

// ---------------------------------------------------------------------------
// Tiny precompute: W1b = bf16(wih @ We) [384,128], bvec = wih @ eb [384] f32
// ---------------------------------------------------------------------------
__global__ void compose_kernel(const float* __restrict__ wih, const float* __restrict__ We,
                               const float* __restrict__ eb, u16* __restrict__ W1b,
                               float* __restrict__ bvec) {
  int id = blockIdx.x * blockDim.x + threadIdx.x;
  if (id < 3 * HID * HID) {
    int i = id >> 7, k = id & 127;
    float s = 0.f;
    for (int n = 0; n < HID; ++n) s += wih[i * HID + n] * We[n * HID + k];
    W1b[id] = f2bf(s);
  }
  if (id < 3 * HID) {
    float s = 0.f;
    for (int n = 0; n < HID; ++n) s += wih[id * HID + n] * eb[n];
    bvec[id] = s;
  }
}

// ---------------------------------------------------------------------------
// Fused GRU: gi = agg @ W1^T + deg*bvec + bih ; gh = h @ whh^T + bhh
//            h' = (1-z)*tanh(in + r*hn) + z*h   (r,z = sigmoid)
// Inner loop: pure bf16 loads + 6 WMMA per k-step.
// In-place safe: all reads of Hin/Hb before __syncthreads(), stores after.
// ---------------------------------------------------------------------------
__global__ __launch_bounds__(256) void gru_kernel(
    const u16* __restrict__ AGGb, const u16* __restrict__ Hb,
    const float* __restrict__ Hin,
    const u16* __restrict__ W1b, const u16* __restrict__ Whhb,
    const float* __restrict__ bih, const float* __restrict__ bhh,
    const float* __restrict__ bvec, const float* __restrict__ deg,
    float* __restrict__ Hout, u16* __restrict__ Hbout) {
  const int lane = threadIdx.x & 31;
  const int wv   = threadIdx.x >> 5;
  const int m0   = blockIdx.x << 4;
  const int n0   = wv << 4;
  const int hi   = lane >> 4;
  const int l15  = lane & 15;
  const size_t aoff = (size_t)(m0 + l15) * HID + (hi << 4);
  const int    col  = n0 + l15;
  const size_t b_r  = (size_t)(col          ) * HID + (hi << 4);
  const size_t b_z  = (size_t)(col + HID    ) * HID + (hi << 4);
  const size_t b_n  = (size_t)(col + 2 * HID) * HID + (hi << 4);

  v8f iR = {}, iZ = {}, iN = {}, hR = {}, hZ = {}, hN = {};
#pragma unroll
  for (int k = 0; k < HID; k += 32) {
    v16bf ax = load_a_bf(AGGb + aoff + k);
    v16bf ah = load_a_bf(Hb + aoff + k);
    v16bf w;
    w = load_b_bf(W1b  + b_r + k); iR = WMMA_BF16(ax, w, iR);
    w = load_b_bf(W1b  + b_z + k); iZ = WMMA_BF16(ax, w, iZ);
    w = load_b_bf(W1b  + b_n + k); iN = WMMA_BF16(ax, w, iN);
    w = load_b_bf(Whhb + b_r + k); hR = WMMA_BF16(ah, w, hR);
    w = load_b_bf(Whhb + b_z + k); hZ = WMMA_BF16(ah, w, hZ);
    w = load_b_bf(Whhb + b_n + k); hN = WMMA_BF16(ah, w, hN);
  }

  const float bi_r = bih[col], bi_z = bih[col + HID], bi_n = bih[col + 2 * HID];
  const float bh_r = bhh[col], bh_z = bhh[col + HID], bh_n = bhh[col + 2 * HID];
  const float bv_r = bvec[col], bv_z = bvec[col + HID], bv_n = bvec[col + 2 * HID];
  const int rbase = m0 + (hi << 3);

  float hold[8], dgv[8];
#pragma unroll
  for (int r = 0; r < 8; ++r) {
    hold[r] = Hin[(size_t)(rbase + r) * HID + col];
    dgv[r]  = deg[rbase + r];
  }
  __syncthreads();  // all Hin/Hb reads complete before any in-place store

#pragma unroll
  for (int r = 0; r < 8; ++r) {
    const float dg  = dgv[r];
    const float sr  = (iR[r] + bi_r + dg * bv_r) + (hR[r] + bh_r);
    const float sz  = (iZ[r] + bi_z + dg * bv_z) + (hZ[r] + bh_z);
    const float hn  = hN[r] + bh_n;
    const float inn = iN[r] + bi_n + dg * bv_n;
    const float rg  = 1.f / (1.f + __expf(-sr));
    const float zg  = 1.f / (1.f + __expf(-sz));
    const float t   = __expf(2.f * (inn + rg * hn));
    const float ng  = (t - 1.f) / (t + 1.f);
    const float hv  = (1.f - zg) * ng + zg * hold[r];
    const size_t o  = (size_t)(rbase + r) * HID + col;
    Hout[o]  = hv;
    Hbout[o] = f2bf(hv);
  }
}

// ---------------------------------------------------------------------------
// Host launch
// ---------------------------------------------------------------------------
extern "C" void kernel_launch(void* const* d_in, const int* in_sizes, int n_in,
                              void* d_out, int out_size, void* d_ws, size_t ws_size,
                              hipStream_t stream) {
  const float* x     = (const float*)d_in[0];
  const int*   eidx  = (const int*)d_in[1];
  const int*   src   = eidx;
  const int*   dst   = eidx + N_EDGES;
  const float* wproj = (const float*)d_in[3];
  const float* bproj = (const float*)d_in[4];
  const float* we    = (const float*)d_in[5];   // [2][1][128][128]
  const float* eb    = (const float*)d_in[6];   // [2][1][128]
  const float* wih   = (const float*)d_in[7];   // [2][384][128]
  const float* whh   = (const float*)d_in[8];   // [2][384][128]
  const float* bih   = (const float*)d_in[9];   // [2][384]
  const float* bhh   = (const float*)d_in[10];  // [2][384]
  float*       out   = (float*)d_out;

  const size_t NH = (size_t)N_NODES * HID;
  char* w = (char*)d_ws;
  float* h    = (float*)w;  w += NH * 4;                          // 51.2 MB
  float* agg  = (float*)w;  w += NH * 4;                          // 51.2 MB
  u16*   Xb   = (u16*)w;    w += (size_t)N_NODES * FEAT * 2;      // 51.2 MB
  u16*   hb   = (u16*)w;    w += NH * 2;                          // 25.6 MB
  u16*   aggb = (u16*)w;    w += NH * 2;                          // 25.6 MB
  u16*   wpb  = (u16*)w;    w += (size_t)HID * FEAT * 2;
  u16*   W1b  = (u16*)w;    w += (size_t)2 * 3 * HID * HID * 2;
  u16*   whhb = (u16*)w;    w += (size_t)2 * 3 * HID * HID * 2;
  float* deg  = (float*)w;  w += (size_t)N_NODES * 4;
  float* bvec = (float*)w;  w += (size_t)2 * 3 * HID * 4;

  // one-time conversions / precomputes (all tiny except Xb)
  cvt_kernel<<<(N_NODES * FEAT / 4 + 255) / 256, 256, 0, stream>>>(x, Xb, N_NODES * FEAT / 4);
  cvt_kernel<<<(HID * FEAT / 4 + 255) / 256, 256, 0, stream>>>(wproj, wpb, HID * FEAT / 4);
  cvt_kernel<<<(2 * 3 * HID * HID / 4 + 255) / 256, 256, 0, stream>>>(whh, whhb, 2 * 3 * HID * HID / 4);
  for (int l = 0; l < 2; ++l) {
    compose_kernel<<<(3 * HID * HID + 255) / 256, 256, 0, stream>>>(
        wih + (size_t)l * 3 * HID * HID, we + (size_t)l * HID * HID,
        eb + (size_t)l * HID, W1b + (size_t)l * 3 * HID * HID, bvec + (size_t)l * 3 * HID);
  }
  zero_kernel<<<(N_NODES + 255) / 256, 256, 0, stream>>>(deg, N_NODES);
  degree_kernel<<<(N_EDGES + 255) / 256, 256, 0, stream>>>(dst, deg, N_EDGES);

  // projection + relu
  proj_kernel<<<N_NODES / 16, 256, 0, stream>>>(Xb, wpb, bproj, h, hb);

  // two GGNN layers
  for (int l = 0; l < 2; ++l) {
    zero_kernel<<<((int)NH + 255) / 256, 256, 0, stream>>>(agg, (int)NH);
    scatter_kernel<<<N_EDGES / 8, 256, 0, stream>>>(h, src, dst, agg, N_EDGES);
    cvt_kernel<<<((int)NH / 4 + 255) / 256, 256, 0, stream>>>(agg, aggb, (int)NH / 4);
    gru_kernel<<<N_NODES / 16, 256, 0, stream>>>(
        aggb, hb, h,
        W1b + (size_t)l * 3 * HID * HID, whhb + (size_t)l * 3 * HID * HID,
        bih + (size_t)l * 3 * HID, bhh + (size_t)l * 3 * HID,
        bvec + (size_t)l * 3 * HID, deg,
        (l == 1) ? out : h, hb);
  }
}
